// DecoderLayer_77988016160858
// MI455X (gfx1250) — matrix operand order repaired and checked by reading, not compiled, not run on hardware
//
#include <hip/hip_runtime.h>

// ---------------------------------------------------------------------------
// Decoder layer for MI455X (gfx1250): all matmuls via v_wmma_f32_16x16x32_bf16
// ---------------------------------------------------------------------------

typedef unsigned short u16;
typedef __attribute__((ext_vector_type(16))) __bf16 v16bf;
typedef __attribute__((ext_vector_type(8)))  float  v8f;

union BF16Frag {
  v16bf v;
  uint4 q[2];
  unsigned int u[8];
};

#define DMODEL 1024
#define NHEADS 16
#define DHEAD  64
#define TLEN   2048
#define BATCH  4
#define MTOK   (BATCH * TLEN)   // 8192 token rows

__device__ __forceinline__ u16 f32_to_bf16(float f) {
  unsigned int x = __float_as_uint(f);
  x += 0x7FFFu + ((x >> 16) & 1u);   // round-to-nearest-even
  return (u16)(x >> 16);
}

__device__ __forceinline__ v8f wmma_bf16(const BF16Frag& a, const BF16Frag& b, v8f c) {
  return __builtin_amdgcn_wmma_f32_16x16x32_bf16(false, a.v, false, b.v,
                                                 (short)0, c, false, false);
}

// ---------------------------------------------------------------------------
// Weight conversion: f32 -> bf16, transposed into Bt ([N][K]) layout
// ---------------------------------------------------------------------------

// wq/wk/wv are [H, D, Dh] f32. Output: wt[(which*1024 + h*64 + e) * 1024 + d]
__global__ __launch_bounds__(256) void conv_qkv_w(const float* __restrict__ wq,
                                                  const float* __restrict__ wk,
                                                  const float* __restrict__ wv,
                                                  u16* __restrict__ wt) {
  size_t idx = (size_t)blockIdx.x * 256 + threadIdx.x;   // 3 * 1024 * 1024 total
  int which = (int)(idx >> 20);
  int rem   = (int)(idx & ((1u << 20) - 1));
  int r = rem >> 10;          // h*64 + e
  int d = rem & 1023;
  int h = r >> 6, e = r & 63;
  const float* src = (which == 0) ? wq : (which == 1) ? wk : wv;
  float v = src[((size_t)h * DMODEL + d) * DHEAD + e];
  wt[idx] = f32_to_bf16(v);
}

// w is [K][N] f32 row-major; wt[n*K + k] = bf16(w[k*N + n])
__global__ __launch_bounds__(256) void conv_t_w(const float* __restrict__ w,
                                                u16* __restrict__ wt,
                                                int K, int N) {
  size_t idx = (size_t)blockIdx.x * 256 + threadIdx.x;
  if (idx >= (size_t)K * N) return;
  int n = (int)(idx / K);
  int k = (int)(idx % K);
  wt[idx] = f32_to_bf16(w[(size_t)k * N + n]);
}

// ---------------------------------------------------------------------------
// LayerNorm: f32 row -> bf16 row
// ---------------------------------------------------------------------------
__global__ __launch_bounds__(256) void ln_kernel(const float* __restrict__ x,
                                                 const float* __restrict__ g,
                                                 const float* __restrict__ bta,
                                                 u16* __restrict__ out, int D) {
  int row = blockIdx.x;
  const float* xr = x + (size_t)row * D;
  float s = 0.f, s2 = 0.f;
  for (int i = threadIdx.x; i < D; i += 256) {
    float v = xr[i];
    s += v; s2 += v * v;
  }
  __shared__ float red[256], red2[256];
  red[threadIdx.x] = s; red2[threadIdx.x] = s2;
  __syncthreads();
  for (int off = 128; off > 0; off >>= 1) {
    if ((int)threadIdx.x < off) {
      red[threadIdx.x]  += red[threadIdx.x + off];
      red2[threadIdx.x] += red2[threadIdx.x + off];
    }
    __syncthreads();
  }
  float mu  = red[0] / (float)D;
  float var = red2[0] / (float)D - mu * mu;
  float rs  = rsqrtf(var + 1e-5f);
  u16* orow = out + (size_t)row * D;
  for (int i = threadIdx.x; i < D; i += 256) {
    float v = (xr[i] - mu) * rs * g[i] + bta[i];
    orow[i] = f32_to_bf16(v);
  }
}

// ---------------------------------------------------------------------------
// Generic bf16 WMMA GEMM: C[M,N] = A[M,K] * Bt[N,K]^T
//   Block: 256 threads (8 waves), tile 128(M) x 256(N), K-step 32.
//   Waves arranged 2(M) x 4(N); each wave computes 64x64 = 4x4 WMMA tiles
//   => 16 v_wmma per K-step from 8 operand fragments (16 ds_load_b128).
// Epilogues: 0 = bf16 store; 1 = bf16 gelu(c + bias); 2 = f32 c + bias + res
// ---------------------------------------------------------------------------
template <int EPI>
__global__ __launch_bounds__(256) void gemm_bf16(const u16* __restrict__ A,
                                                 const u16* __restrict__ Bt,
                                                 u16* __restrict__ outB,
                                                 float* __restrict__ outF,
                                                 const float* __restrict__ bias,
                                                 const float* __restrict__ res,
                                                 int N, int K) {
  __shared__ u16 As[128][40];   // 32 K cols + pad, row stride 80 B (16B-aligned)
  __shared__ u16 Bs[256][40];

  const int bm = blockIdx.y * 128;
  const int bn = blockIdx.x * 256;
  const int wave = threadIdx.x >> 5;
  const int lane = threadIdx.x & 31;
  const int wm = wave >> 2;       // 0..1  (M)
  const int wn = wave & 3;        // 0..3  (N)
  const int lh = lane & 15;
  const int hi = lane >> 4;       // 0 or 1
  const int ab = hi ? 8 : 0;      // A-operand K sub-base
  const int bb = hi ? 8 : 0;      // B-operand K base

  v8f acc[4][4];
#pragma unroll
  for (int i = 0; i < 4; ++i)
#pragma unroll
    for (int j = 0; j < 4; ++j)
#pragma unroll
      for (int r = 0; r < 8; ++r) acc[i][j][r] = 0.f;

  const int sr = threadIdx.x >> 2;          // 0..63
  const int sc = (threadIdx.x & 3) * 8;     // 0,8,16,24

  for (int k0 = 0; k0 < K; k0 += 32) {
    // Stage current K-slab into registers (overlaps trailing compute).
    uint4 va0 = *(const uint4*)(A  + (size_t)(bm + sr)      * K + k0 + sc);
    uint4 va1 = *(const uint4*)(A  + (size_t)(bm + sr + 64) * K + k0 + sc);
    uint4 vb0 = *(const uint4*)(Bt + (size_t)(bn + sr)       * K + k0 + sc);
    uint4 vb1 = *(const uint4*)(Bt + (size_t)(bn + sr + 64)  * K + k0 + sc);
    uint4 vb2 = *(const uint4*)(Bt + (size_t)(bn + sr + 128) * K + k0 + sc);
    uint4 vb3 = *(const uint4*)(Bt + (size_t)(bn + sr + 192) * K + k0 + sc);
    // Run the global pipe ahead of next iteration (global_prefetch_b8).
    if (k0 + 32 < K) {
      __builtin_prefetch(A  + (size_t)(bm + sr)       * K + k0 + 32 + sc, 0, 1);
      __builtin_prefetch(A  + (size_t)(bm + sr + 64)  * K + k0 + 32 + sc, 0, 1);
      __builtin_prefetch(Bt + (size_t)(bn + sr)       * K + k0 + 32 + sc, 0, 1);
      __builtin_prefetch(Bt + (size_t)(bn + sr + 64)  * K + k0 + 32 + sc, 0, 1);
      __builtin_prefetch(Bt + (size_t)(bn + sr + 128) * K + k0 + 32 + sc, 0, 1);
      __builtin_prefetch(Bt + (size_t)(bn + sr + 192) * K + k0 + 32 + sc, 0, 1);
    }
    __syncthreads();   // previous iteration's LDS reads are done
    *(uint4*)&As[sr][sc]       = va0;
    *(uint4*)&As[sr + 64][sc]  = va1;
    *(uint4*)&Bs[sr][sc]       = vb0;
    *(uint4*)&Bs[sr + 64][sc]  = vb1;
    *(uint4*)&Bs[sr + 128][sc] = vb2;
    *(uint4*)&Bs[sr + 192][sc] = vb3;
    __syncthreads();

    BF16Frag af[4], bf[4];
#pragma unroll
    for (int i = 0; i < 4; ++i) {
      int ar = wm * 64 + i * 16 + lh;
      af[i].q[0] = *(const uint4*)&As[ar][ab];
      af[i].q[1] = *(const uint4*)&As[ar][16 + ab];
      int br = wn * 64 + i * 16 + lh;
      bf[i].q[0] = *(const uint4*)&Bs[br][bb];
      bf[i].q[1] = *(const uint4*)&Bs[br][16 + bb];
    }
#pragma unroll
    for (int mi = 0; mi < 4; ++mi)
#pragma unroll
      for (int ni = 0; ni < 4; ++ni)
        acc[mi][ni] = wmma_bf16(af[mi], bf[ni], acc[mi][ni]);
  }

  // Epilogue. D layout: lane<16 -> M = r, lane>=16 -> M = r + 8; N = lane%16.
#pragma unroll
  for (int mi = 0; mi < 4; ++mi) {
#pragma unroll
    for (int ni = 0; ni < 4; ++ni) {
      int col = bn + wn * 64 + ni * 16 + lh;
#pragma unroll
      for (int r = 0; r < 8; ++r) {
        int row = bm + wm * 64 + mi * 16 + r + hi * 8;
        float v = acc[mi][ni][r];
        if (EPI == 0) {
          outB[(size_t)row * N + col] = f32_to_bf16(v);
        } else if (EPI == 1) {
          float t = v + bias[col];
          float gl = 0.5f * t * (1.f + erff(t * 0.70710678118654752f));
          outB[(size_t)row * N + col] = f32_to_bf16(gl);
        } else {
          outF[(size_t)row * N + col] = v + bias[col] + res[(size_t)row * N + col];
        }
      }
    }
  }
}

// ---------------------------------------------------------------------------
// Flash attention, causal. qkv layout: [B*T, 3072] bf16 (Q | K | V per head).
// Block: 128 threads (4 waves), grid (T/64, H, B). Wave w handles 16 q-rows.
// Output: [B*T, 1024] bf16 with head-concatenated layout (== transpose+reshape)
// ---------------------------------------------------------------------------
__global__ __launch_bounds__(128) void attn_kernel(const u16* __restrict__ qkv,
                                                   u16* __restrict__ out) {
  const int LDQ = 3 * DMODEL;   // 3072
  const int qb = blockIdx.x;    // 0..31
  const int h  = blockIdx.y;
  const int b  = blockIdx.z;
  const int wave = threadIdx.x >> 5;
  const int lane = threadIdx.x & 31;
  const int lh = lane & 15;
  const int hi = lane >> 4;
  const int ab = hi ? 8 : 0;
  const int kb = hi ? 8 : 0;
  const float scale = 0.125f;   // 1/sqrt(64)

  __shared__ u16 Ks[64][72];        // [s][e]
  __shared__ u16 Vs[64][72];        // [e][s] (transposed at stage time)
  __shared__ u16 Ps[4][16][72];     // per-wave P scratch (D-layout -> A-layout)

  // --- Load Q fragments (A-operand layout) once ---
  const int m0 = qb * 64 + wave * 16;
  const u16* qptr = qkv + (size_t)(b * TLEN + m0 + lh) * LDQ + h * DHEAD;
  BF16Frag aq0, aq1;
#pragma unroll
  for (int i = 0; i < 4; ++i) {
    aq0.u[i]     = *(const unsigned int*)(qptr + ab + 2 * i);
    aq0.u[4 + i] = *(const unsigned int*)(qptr + 16 + ab + 2 * i);
    aq1.u[i]     = *(const unsigned int*)(qptr + 32 + ab + 2 * i);
    aq1.u[4 + i] = *(const unsigned int*)(qptr + 48 + ab + 2 * i);
  }

  float Mrun[8], Srun[8];
  v8f accO[4];
#pragma unroll
  for (int r = 0; r < 8; ++r) { Mrun[r] = -1e30f; Srun[r] = 0.f; }
#pragma unroll
  for (int n = 0; n < 4; ++n)
#pragma unroll
    for (int r = 0; r < 8; ++r) accO[n][r] = 0.f;

  const int nSB = qb + 1;   // causal: s-blocks 0..qb
  for (int sb = 0; sb < nSB; ++sb) {
    const int s0 = sb * 64;
    __syncthreads();   // previous iteration's Ks/Vs reads complete
    // Stage K tile [s][e]
    for (int idx = threadIdx.x; idx < 64 * 8; idx += 128) {
      int sl = idx >> 3, c8 = (idx & 7) * 8;
      uint4 kv = *(const uint4*)(qkv + (size_t)(b * TLEN + s0 + sl) * LDQ +
                                 DMODEL + h * DHEAD + c8);
      *(uint4*)&Ks[sl][c8] = kv;
    }
    // Stage V transposed: Vs[e][s]
    for (int idx = threadIdx.x; idx < 64 * 32; idx += 128) {
      int sl = idx >> 5, e2 = (idx & 31) * 2;
      unsigned int vv = *(const unsigned int*)(qkv + (size_t)(b * TLEN + s0 + sl) * LDQ +
                                               2 * DMODEL + h * DHEAD + e2);
      Vs[e2][sl]     = (u16)(vv & 0xffffu);
      Vs[e2 + 1][sl] = (u16)(vv >> 16);
    }
    __syncthreads();

    // --- S = Q * K^T (16 x 64 per wave) ---
    v8f accS[4];
#pragma unroll
    for (int ns = 0; ns < 4; ++ns) {
      BF16Frag b0, b1;
      int srow = ns * 16 + lh;
      b0.q[0] = *(const uint4*)&Ks[srow][kb];
      b0.q[1] = *(const uint4*)&Ks[srow][16 + kb];
      b1.q[0] = *(const uint4*)&Ks[srow][32 + kb];
      b1.q[1] = *(const uint4*)&Ks[srow][48 + kb];
      v8f s;
#pragma unroll
      for (int r = 0; r < 8; ++r) s[r] = 0.f;
      s = wmma_bf16(aq0, b0, s);
      s = wmma_bf16(aq1, b1, s);
      accS[ns] = s;
    }

    // --- mask + scale, block row-max ---
    float bmax[8];
#pragma unroll
    for (int r = 0; r < 8; ++r) bmax[r] = -1e30f;
#pragma unroll
    for (int ns = 0; ns < 4; ++ns) {
      int sg = s0 + ns * 16 + lh;
#pragma unroll
      for (int r = 0; r < 8; ++r) {
        int mg = m0 + r + hi * 8;
        float v = accS[ns][r] * scale;
        if (sg > mg) v = -1e30f;
        accS[ns][r] = v;
        bmax[r] = fmaxf(bmax[r], v);
      }
    }
#pragma unroll
    for (int off = 1; off < 16; off <<= 1) {
#pragma unroll
      for (int r = 0; r < 8; ++r)
        bmax[r] = fmaxf(bmax[r], __shfl_xor(bmax[r], off, 32));
    }

    // --- online softmax update ---
    float alpha[8];
#pragma unroll
    for (int r = 0; r < 8; ++r) {
      float mnew = fmaxf(Mrun[r], bmax[r]);
      alpha[r] = __expf(Mrun[r] - mnew);
      Mrun[r] = mnew;
    }
    float bsum[8];
#pragma unroll
    for (int r = 0; r < 8; ++r) bsum[r] = 0.f;
#pragma unroll
    for (int ns = 0; ns < 4; ++ns)
#pragma unroll
      for (int r = 0; r < 8; ++r) {
        float p = __expf(accS[ns][r] - Mrun[r]);
        accS[ns][r] = p;
        bsum[r] += p;
      }
#pragma unroll
    for (int off = 1; off < 16; off <<= 1) {
#pragma unroll
      for (int r = 0; r < 8; ++r)
        bsum[r] += __shfl_xor(bsum[r], off, 32);
    }
#pragma unroll
    for (int r = 0; r < 8; ++r) Srun[r] = Srun[r] * alpha[r] + bsum[r];
#pragma unroll
    for (int n = 0; n < 4; ++n)
#pragma unroll
      for (int r = 0; r < 8; ++r) accO[n][r] *= alpha[r];

    // --- write P (D-layout) to LDS scratch ---
#pragma unroll
    for (int ns = 0; ns < 4; ++ns)
#pragma unroll
      for (int r = 0; r < 8; ++r)
        Ps[wave][r + hi * 8][ns * 16 + lh] = f32_to_bf16(accS[ns][r]);
    __syncthreads();

    // --- re-read P as A operand, then O += P * V ---
    BF16Frag ap0, ap1;
    ap0.q[0] = *(const uint4*)&Ps[wave][lh][ab];
    ap0.q[1] = *(const uint4*)&Ps[wave][lh][16 + ab];
    ap1.q[0] = *(const uint4*)&Ps[wave][lh][32 + ab];
    ap1.q[1] = *(const uint4*)&Ps[wave][lh][48 + ab];
#pragma unroll
    for (int ne = 0; ne < 4; ++ne) {
      BF16Frag b0, b1;
      int erow = ne * 16 + lh;
      b0.q[0] = *(const uint4*)&Vs[erow][kb];
      b0.q[1] = *(const uint4*)&Vs[erow][16 + kb];
      b1.q[0] = *(const uint4*)&Vs[erow][32 + kb];
      b1.q[1] = *(const uint4*)&Vs[erow][48 + kb];
      accO[ne] = wmma_bf16(ap0, b0, accO[ne]);
      accO[ne] = wmma_bf16(ap1, b1, accO[ne]);
    }
  }

  // --- epilogue: out[b, m, h*64 + e] = O / Srun ---
#pragma unroll
  for (int ne = 0; ne < 4; ++ne) {
    int e = ne * 16 + lh;
#pragma unroll
    for (int r = 0; r < 8; ++r) {
      int row = m0 + r + hi * 8;
      float v = accO[ne][r] / Srun[r];
      out[(size_t)(b * TLEN + row) * DMODEL + h * DHEAD + e] = f32_to_bf16(v);
    }
  }
}

// ---------------------------------------------------------------------------
// Launcher
// ---------------------------------------------------------------------------
extern "C" void kernel_launch(void* const* d_in, const int* in_sizes, int n_in,
                              void* d_out, int out_size, void* d_ws, size_t ws_size,
                              hipStream_t stream) {
  const float* x    = (const float*)d_in[0];
  const float* g1   = (const float*)d_in[1];
  const float* bt1  = (const float*)d_in[2];
  const float* wq   = (const float*)d_in[3];
  const float* wk   = (const float*)d_in[4];
  const float* wv   = (const float*)d_in[5];
  const float* wo   = (const float*)d_in[6];
  const float* bo   = (const float*)d_in[7];
  const float* g2   = (const float*)d_in[8];
  const float* bt2  = (const float*)d_in[9];
  const float* wfc  = (const float*)d_in[10];
  const float* bfc  = (const float*)d_in[11];
  const float* wpr  = (const float*)d_in[12];
  const float* bpr  = (const float*)d_in[13];
  float* out = (float*)d_out;

  char* wsb = (char*)d_ws;
  size_t off = 0;
  auto alloc = [&](size_t bytes) -> char* {
    char* p = wsb + off;
    off += (bytes + 255) & ~(size_t)255;
    return p;
  };
  u16*   hbuf  = (u16*)alloc((size_t)MTOK * DMODEL * 2);       // ln1 out (bf16)
  u16*   qkvb  = (u16*)alloc((size_t)MTOK * 3 * DMODEL * 2);   // q|k|v (bf16)
  u16*   attno = (u16*)alloc((size_t)MTOK * DMODEL * 2);       // attn out (bf16)
  float* x1    = (float*)alloc((size_t)MTOK * DMODEL * 4);     // residual 1 (f32)
  u16*   h2    = (u16*)alloc((size_t)MTOK * DMODEL * 2);       // ln2 out (bf16)
  u16*   mbuf  = (u16*)alloc((size_t)MTOK * 4 * DMODEL * 2);   // gelu(fc) (bf16)
  u16*   wqkvt = (u16*)alloc((size_t)3 * DMODEL * DMODEL * 2); // [3072][1024]
  u16*   wot   = (u16*)alloc((size_t)DMODEL * DMODEL * 2);     // [1024][1024]
  u16*   wfct  = (u16*)alloc((size_t)4 * DMODEL * DMODEL * 2); // [4096][1024]
  u16*   wprt  = (u16*)alloc((size_t)DMODEL * 4 * DMODEL * 2); // [1024][4096]
  (void)in_sizes; (void)n_in; (void)out_size; (void)ws_size;

  // Weight conversion (cheap vs matmuls; deterministic each call)
  conv_qkv_w<<<(3 * 1024 * 1024) / 256, 256, 0, stream>>>(wq, wk, wv, wqkvt);
  conv_t_w<<<(1024 * 1024) / 256, 256, 0, stream>>>(wo, wot, 1024, 1024);
  conv_t_w<<<(1024 * 4096) / 256, 256, 0, stream>>>(wfc, wfct, 1024, 4096);
  conv_t_w<<<(4096 * 1024) / 256, 256, 0, stream>>>(wpr, wprt, 4096, 1024);

  // h = LN1(x)  (bf16)
  ln_kernel<<<MTOK, 256, 0, stream>>>(x, g1, bt1, hbuf, DMODEL);

  // qkv = h @ Wqkv   [8192 x 3072]
  gemm_bf16<0><<<dim3(3072 / 256, MTOK / 128), 256, 0, stream>>>(
      hbuf, wqkvt, qkvb, nullptr, nullptr, nullptr, 3072, 1024);

  // attention (flash, causal)
  attn_kernel<<<dim3(TLEN / 64, NHEADS, BATCH), 128, 0, stream>>>(qkvb, attno);

  // x1 = x + attn_out @ Wo + bo   (f32)
  gemm_bf16<2><<<dim3(1024 / 256, MTOK / 128), 256, 0, stream>>>(
      attno, wot, nullptr, x1, bo, x, 1024, 1024);

  // h2 = LN2(x1)  (bf16)
  ln_kernel<<<MTOK, 256, 0, stream>>>(x1, g2, bt2, h2, DMODEL);

  // m = gelu(h2 @ Wfc + bfc)   [8192 x 4096] bf16
  gemm_bf16<1><<<dim3(4096 / 256, MTOK / 128), 256, 0, stream>>>(
      h2, wfct, mbuf, nullptr, bfc, nullptr, 4096, 1024);

  // out = x1 + m @ Wpr + bpr   (f32)
  gemm_bf16<2><<<dim3(1024 / 256, MTOK / 128), 256, 0, stream>>>(
      mbuf, wprt, nullptr, out, bpr, x1, 1024, 4096);
}
